// SSClassifier_VanBerg_Longformer_10651518894348
// MI455X (gfx1250) — matrix operand order repaired and checked by reading, not compiled, not run on hardware
//
#include <hip/hip_runtime.h>

// ---------------- problem constants ----------------
#define B_     2
#define S_     4096
#define D_     768
#define H_     12
#define DH_    64
#define L_     2
#define W_     256
#define NB_    16          // S/W
#define G_     64
#define FF_    3072
#define C_     7
#define NSEPS_ 63
#define SEP_   2
#define NEGV   (-1.0e9f)
#define MTOK   (B_*S_)     // 8192

typedef __attribute__((ext_vector_type(16))) _Float16 v16h;
typedef __attribute__((ext_vector_type(8)))  float    v8f;

// ---- optional gfx1250 async global->LDS path (guarded; falls back to ld/st) ----
#if defined(__HIP_DEVICE_COMPILE__) && defined(__gfx1250__) && \
    __has_builtin(__builtin_amdgcn_global_load_async_to_lds_b128)
#define ASYNC_LDS 1
typedef int vi4_t __attribute__((vector_size(16)));
typedef __attribute__((address_space(1))) vi4_t* g4p_t;
typedef __attribute__((address_space(3))) vi4_t* l4p_t;
__device__ __forceinline__ void async_cp16(const void* g, void* l) {
  __builtin_amdgcn_global_load_async_to_lds_b128((g4p_t)g, (l4p_t)l, 0, 0);
}
__device__ __forceinline__ void async_wait0() {
#if __has_builtin(__builtin_amdgcn_s_wait_asynccnt)
  __builtin_amdgcn_s_wait_asynccnt(0);
#else
  asm volatile("s_wait_asynccnt 0" ::: "memory");
#endif
}
#else
#define ASYNC_LDS 0
#endif

__device__ __forceinline__ v8f wmma32(v16h a, v16h b, v8f c) {
  return __builtin_amdgcn_wmma_f32_16x16x32_f16(false, a, false, b, (short)0, c, false, false);
}

__device__ __forceinline__ v8f zero8() {
  v8f r;
#pragma unroll
  for (int i = 0; i < 8; ++i) r[i] = 0.f;
  return r;
}

// load a 16-element f16 fragment as two 16B chunks
__device__ __forceinline__ v16h ldfrag(const _Float16* p0, const _Float16* p1) {
  v16h r;
  uint4* q = reinterpret_cast<uint4*>(&r);
  q[0] = *reinterpret_cast<const uint4*>(p0);
  q[1] = *reinterpret_cast<const uint4*>(p1);
  return r;
}

__device__ __forceinline__ float shflxor(float x, int m) { return __shfl_xor(x, m, 32); }

__device__ __forceinline__ float gelu_f(float x) {
  float x3 = x * x * x;
  return 0.5f * x * (1.f + tanhf(0.7978845608028654f * (x + 0.044715f * x3)));
}

// ---------------- f32 -> f16 convert ----------------
__global__ void k_f2h(const float* __restrict__ src, _Float16* __restrict__ dst, long long n) {
  long long i = (long long)blockIdx.x * blockDim.x + threadIdx.x;
  if (i < n) dst[i] = (_Float16)src[i];
}

// ---------------- global-token index + key_local_ok ----------------
__global__ void k_globals(const int* __restrict__ ids, const int* __restrict__ am,
                          int* __restrict__ idx_g, int* __restrict__ klo) {
  int b = blockIdx.x;
  int tid = threadIdx.x;
  if (tid == 0) {
    int cnt = 0;
    for (int s = 0; s < S_; ++s)
      if ((ids[b * S_ + s] == SEP_ || s == 0) && cnt < G_) idx_g[b * G_ + cnt++] = s;
    for (; cnt < G_; ++cnt) idx_g[b * G_ + cnt] = 0;
  }
  for (int s = tid; s < S_; s += blockDim.x) {
    bool g = (ids[b * S_ + s] == SEP_) || (s == 0);
    klo[b * S_ + s] = (am[b * S_ + s] != 0 && !g) ? 1 : 0;
  }
}

// ---------------- embedding + LayerNorm ----------------
__global__ __launch_bounds__(256) void k_embed_ln(const float* __restrict__ emb_tok,
                                                  const float* __restrict__ emb_pos,
                                                  const float* __restrict__ ls,
                                                  const float* __restrict__ lb,
                                                  const int* __restrict__ ids,
                                                  float* __restrict__ xo, _Float16* __restrict__ xh) {
  int bid = blockIdx.x;
  int b = bid / S_, s = bid % S_;
  int tid = threadIdx.x;
  __shared__ float red[256];
  int tok = ids[b * S_ + s];
  float v[3];
#pragma unroll
  for (int kk = 0; kk < 3; ++kk) {
    int d = tid + kk * 256;
    v[kk] = emb_tok[(size_t)tok * D_ + d] + emb_pos[(size_t)s * D_ + d];
  }
  float sum = v[0] + v[1] + v[2];
  red[tid] = sum; __syncthreads();
  for (int off = 128; off > 0; off >>= 1) { if (tid < off) red[tid] += red[tid + off]; __syncthreads(); }
  float mean = red[0] * (1.f / D_);
  __syncthreads();
  float s2 = 0.f;
#pragma unroll
  for (int kk = 0; kk < 3; ++kk) { float d = v[kk] - mean; s2 += d * d; }
  red[tid] = s2; __syncthreads();
  for (int off = 128; off > 0; off >>= 1) { if (tid < off) red[tid] += red[tid + off]; __syncthreads(); }
  float rs = rsqrtf(red[0] * (1.f / D_) + 1e-5f);
  size_t base = (size_t)bid * D_;
#pragma unroll
  for (int kk = 0; kk < 3; ++kk) {
    int d = tid + kk * 256;
    float o = (v[kk] - mean) * rs * ls[d] + lb[d];
    xo[base + d] = o;
    xh[base + d] = (_Float16)o;
  }
}

// ---------------- residual + LayerNorm ----------------
__global__ __launch_bounds__(256) void k_ln_resid(const float* __restrict__ xin,
                                                  const float* __restrict__ p,
                                                  const float* __restrict__ ls,
                                                  const float* __restrict__ lb,
                                                  float* __restrict__ xo, _Float16* __restrict__ xh) {
  int bid = blockIdx.x;
  int tid = threadIdx.x;
  __shared__ float red[256];
  size_t base = (size_t)bid * D_;
  float v[3];
#pragma unroll
  for (int kk = 0; kk < 3; ++kk) {
    int d = tid + kk * 256;
    v[kk] = xin[base + d] + p[base + d];
  }
  float sum = v[0] + v[1] + v[2];
  red[tid] = sum; __syncthreads();
  for (int off = 128; off > 0; off >>= 1) { if (tid < off) red[tid] += red[tid + off]; __syncthreads(); }
  float mean = red[0] * (1.f / D_);
  __syncthreads();
  float s2 = 0.f;
#pragma unroll
  for (int kk = 0; kk < 3; ++kk) { float d = v[kk] - mean; s2 += d * d; }
  red[tid] = s2; __syncthreads();
  for (int off = 128; off > 0; off >>= 1) { if (tid < off) red[tid] += red[tid + off]; __syncthreads(); }
  float rs = rsqrtf(red[0] * (1.f / D_) + 1e-5f);
#pragma unroll
  for (int kk = 0; kk < 3; ++kk) {
    int d = tid + kk * 256;
    float o = (v[kk] - mean) * rs * ls[d] + lb[d];
    xo[base + d] = o;
    xh[base + d] = (_Float16)o;
  }
}

// ---------------- WMMA GEMM: Y[M,N] = A[M,K] @ B[K,N] + bias, opt GELU ----------------
// M % 128 == 0, N % 64 == 0, K % 32 == 0
__global__ __launch_bounds__(256) void k_gemm(const _Float16* __restrict__ A,
                                              const _Float16* __restrict__ Bw,
                                              const float* __restrict__ bias,
                                              float* __restrict__ outF, _Float16* __restrict__ outH,
                                              int M, int N, int K, int act) {
  __shared__ __align__(16) _Float16 As[128][32];
  __shared__ __align__(16) _Float16 Bs[64][32]; // stored transposed: Bs[n][k]
  int tid  = threadIdx.x;
  int wave = tid >> 5, lane = tid & 31, half = lane >> 4, lm = lane & 15;
  int wr = wave & 3, wc = wave >> 2;
  int bm = blockIdx.y * 128, bn = blockIdx.x * 64;
  v8f acc[2][2];
  acc[0][0] = zero8(); acc[0][1] = zero8(); acc[1][0] = zero8(); acc[1][1] = zero8();
  int arow = tid >> 1, aseg = tid & 1;
  int bk = tid >> 3, bnn = (tid & 7) * 8;
  for (int k0 = 0; k0 < K; k0 += 32) {
    // stage A tile (row-major 128x32)
    const _Float16* asrc = A + (size_t)(bm + arow) * K + k0 + aseg * 16;
    _Float16* adst = &As[arow][aseg * 16];
#if ASYNC_LDS
    async_cp16(asrc, adst);
    async_cp16(asrc + 8, adst + 8);
#else
    {
      const uint4* ap = reinterpret_cast<const uint4*>(asrc);
      uint4 a0 = ap[0], a1 = ap[1];
      uint4* ad = reinterpret_cast<uint4*>(adst);
      ad[0] = a0; ad[1] = a1;
    }
#endif
    // stage B tile transposed (Bs[n][k])
    uint4 bvv = *reinterpret_cast<const uint4*>(Bw + (size_t)(k0 + bk) * N + bn + bnn);
    const _Float16* be = reinterpret_cast<const _Float16*>(&bvv);
#pragma unroll
    for (int j = 0; j < 8; ++j) Bs[bnn + j][bk] = be[j];
    if (k0 + 32 < K) {
      __builtin_prefetch(A + (size_t)(bm + arow) * K + k0 + 32 + aseg * 16, 0, 0);
      __builtin_prefetch(Bw + (size_t)(k0 + 32 + bk) * N + bn + bnn, 0, 0);
    }
#if ASYNC_LDS
    async_wait0();
#endif
    __syncthreads();
    v16h af[2], bf[2];
#pragma unroll
    for (int r = 0; r < 2; ++r) {
      const _Float16* arp = &As[wr * 32 + r * 16 + lm][half * 8];
      af[r] = ldfrag(arp, arp + 16);
    }
#pragma unroll
    for (int c = 0; c < 2; ++c) {
      const _Float16* brp = &Bs[wc * 32 + c * 16 + lm][half * 8];
      bf[c] = ldfrag(brp, brp + 16);
    }
#pragma unroll
    for (int r = 0; r < 2; ++r)
#pragma unroll
      for (int c = 0; c < 2; ++c) acc[r][c] = wmma32(af[r], bf[c], acc[r][c]);
    __syncthreads();
  }
#pragma unroll
  for (int r = 0; r < 2; ++r)
#pragma unroll
    for (int c = 0; c < 2; ++c) {
      int col = bn + wc * 32 + c * 16 + lm;
      float bi = bias ? bias[col] : 0.f;
#pragma unroll
      for (int j = 0; j < 8; ++j) {
        int row = bm + wr * 32 + r * 16 + j + half * 8;
        float vv = acc[r][c][j] + bi;
        if (act == 1) vv = gelu_f(vv);
        size_t off = (size_t)row * N + col;
        if (outF) outF[off] = vv;
        if (outH) outH[off] = (_Float16)vv;
      }
    }
}

// ---------------- banded local attention (flash-style, WMMA) ----------------
__global__ __launch_bounds__(512) void k_band_attn(const _Float16* __restrict__ q,
                                                   const _Float16* __restrict__ k,
                                                   const _Float16* __restrict__ v,
                                                   const int* __restrict__ idx_g,
                                                   const int* __restrict__ klo,
                                                   float* __restrict__ out) {
  __shared__ __align__(16) _Float16 Ks[32][64];   // [key][dh]
  __shared__ __align__(16) _Float16 Vt[64][32];   // [dh][key]
  __shared__ __align__(16) _Float16 Ps[16][16 * 32];
  int h = blockIdx.x, n = blockIdx.y, b = blockIdx.z;
  int tid = threadIdx.x;
  int wave = tid >> 5, lane = tid & 31, half = lane >> 4, lm = lane & 15;
  const float scale = 0.125f;
  int qbase = n * W_ + wave * 16;
  const _Float16* qp = q + ((size_t)(b * S_) + qbase + lm) * (size_t)D_ + h * DH_;
  v16h qf0 = ldfrag(qp + half * 8, qp + 16 + half * 8);
  v16h qf1 = ldfrag(qp + 32 + half * 8, qp + 48 + half * 8);
  _Float16* Psw = Ps[wave];
  float m_r[8], l_r[8];
  v8f O[4];
#pragma unroll
  for (int j = 0; j < 8; ++j) { m_r[j] = -3.0e38f; l_r[j] = 0.f; }
#pragma unroll
  for (int t = 0; t < 4; ++t) O[t] = zero8();
  int kband0 = n * W_ - W_;
  for (int ch = 0; ch < 26; ++ch) {
    bool isg = (ch >= 24);
    // ---- stage K (threads 0..255) and Vt (threads 256..511) ----
    if (tid < 256) {
      int key = tid >> 3, seg = tid & 7;
      int kp = isg ? idx_g[b * G_ + (ch - 24) * 32 + key] : (kband0 + ch * 32 + key);
#if ASYNC_LDS
      // async path only when the whole chunk is in-range (uniform per chunk)
      bool chunk_ok = isg || ((kband0 + ch * 32) >= 0 && (kband0 + ch * 32 + 32) <= S_);
      if (chunk_ok) {
        async_cp16(k + ((size_t)b * S_ + kp) * D_ + h * DH_ + seg * 8, &Ks[key][seg * 8]);
      } else
#endif
      {
        uint4 val = make_uint4(0, 0, 0, 0);
        if (kp >= 0 && kp < S_)
          val = *reinterpret_cast<const uint4*>(k + ((size_t)b * S_ + kp) * D_ + h * DH_ + seg * 8);
        *reinterpret_cast<uint4*>(&Ks[key][seg * 8]) = val;
      }
    } else {
      int t2 = tid - 256;
      int key = t2 >> 3, seg = t2 & 7;
      int kp = isg ? idx_g[b * G_ + (ch - 24) * 32 + key] : (kband0 + ch * 32 + key);
      uint4 val = make_uint4(0, 0, 0, 0);
      if (kp >= 0 && kp < S_)
        val = *reinterpret_cast<const uint4*>(v + ((size_t)b * S_ + kp) * D_ + h * DH_ + seg * 8);
      const _Float16* e = reinterpret_cast<const _Float16*>(&val);
#pragma unroll
      for (int j = 0; j < 8; ++j) Vt[seg * 8 + j][key] = e[j];
    }
#if ASYNC_LDS
    async_wait0();
#endif
    __syncthreads();
    // ---- scores: S = Q Kt (two 16-key tiles) ----
    v8f s0 = zero8(), s1 = zero8();
    {
      const _Float16* kr0 = &Ks[lm][half * 8];
      const _Float16* kr1 = &Ks[16 + lm][half * 8];
      v16h kf00 = ldfrag(kr0, kr0 + 16);
      v16h kf01 = ldfrag(kr0 + 32, kr0 + 48);
      v16h kf10 = ldfrag(kr1, kr1 + 16);
      v16h kf11 = ldfrag(kr1 + 32, kr1 + 48);
      s0 = wmma32(qf0, kf00, s0); s0 = wmma32(qf1, kf01, s0);
      s1 = wmma32(qf0, kf10, s1); s1 = wmma32(qf1, kf11, s1);
    }
    int kp0 = 0, kp1 = 0; bool ok0 = true, ok1 = true;
    if (!isg) {
      kp0 = kband0 + ch * 32 + lm;
      kp1 = kp0 + 16;
      ok0 = (kp0 >= 0 && kp0 < S_) && (klo[b * S_ + kp0] != 0);
      ok1 = (kp1 >= 0 && kp1 < S_) && (klo[b * S_ + kp1] != 0);
    }
    // ---- online softmax update ----
#pragma unroll
    for (int j = 0; j < 8; ++j) {
      int row = j + half * 8;
      int qpos = qbase + row;
      float sv0 = s0[j] * scale, sv1 = s1[j] * scale;
      if (!isg) {
        int d0 = kp0 - qpos, d1 = kp1 - qpos;
        if (!(ok0 && d0 <= W_ && d0 >= -W_)) sv0 = NEGV;
        if (!(ok1 && d1 <= W_ && d1 >= -W_)) sv1 = NEGV;
      }
      float mx = fmaxf(sv0, sv1);
      mx = fmaxf(mx, shflxor(mx, 1));
      mx = fmaxf(mx, shflxor(mx, 2));
      mx = fmaxf(mx, shflxor(mx, 4));
      mx = fmaxf(mx, shflxor(mx, 8));
      float mnew = fmaxf(m_r[j], mx);
      float fct = __expf(m_r[j] - mnew);
      m_r[j] = mnew;
      float p0 = __expf(sv0 - mnew), p1 = __expf(sv1 - mnew);
      float rs = p0 + p1;
      rs += shflxor(rs, 1); rs += shflxor(rs, 2); rs += shflxor(rs, 4); rs += shflxor(rs, 8);
      l_r[j] = l_r[j] * fct + rs;
      O[0][j] *= fct; O[1][j] *= fct; O[2][j] *= fct; O[3][j] *= fct;
      Psw[row * 32 + lm]      = (_Float16)p0;
      Psw[row * 32 + 16 + lm] = (_Float16)p1;
    }
    asm volatile("s_wait_dscnt 0" ::: "memory");
    // ---- O += P @ V ----
    {
      const _Float16* pr = Psw + lm * 32 + half * 8;
      v16h pf = ldfrag(pr, pr + 16);
#pragma unroll
      for (int tn = 0; tn < 4; ++tn) {
        const _Float16* vr = &Vt[tn * 16 + lm][half * 8];
        v16h vf = ldfrag(vr, vr + 16);
        O[tn] = wmma32(pf, vf, O[tn]);
      }
    }
    __syncthreads();
  }
  // ---- writeback ----
#pragma unroll
  for (int j = 0; j < 8; ++j) {
    int row = j + half * 8;
    int qpos = qbase + row;
    float inv = 1.f / l_r[j];
#pragma unroll
    for (int tn = 0; tn < 4; ++tn)
      out[((size_t)b * S_ + qpos) * D_ + h * DH_ + tn * 16 + lm] = O[tn][j] * inv;
  }
}

// ---------------- global-token attention over all S keys ----------------
__global__ __launch_bounds__(128) void k_glob_attn(const _Float16* __restrict__ qg,
                                                   const _Float16* __restrict__ kk,
                                                   const _Float16* __restrict__ vv,
                                                   const int* __restrict__ am,
                                                   float* __restrict__ og) {
  __shared__ __align__(16) _Float16 Ks[32][64];
  __shared__ __align__(16) _Float16 Vt[64][32];
  __shared__ __align__(16) _Float16 Ps[4][16 * 32];
  int h = blockIdx.x, b = blockIdx.y;
  int tid = threadIdx.x;
  int wave = tid >> 5, lane = tid & 31, half = lane >> 4, lm = lane & 15;
  const float scale = 0.125f;
  const _Float16* qp = qg + ((size_t)b * G_ + wave * 16 + lm) * (size_t)D_ + h * DH_;
  v16h qf0 = ldfrag(qp + half * 8, qp + 16 + half * 8);
  v16h qf1 = ldfrag(qp + 32 + half * 8, qp + 48 + half * 8);
  _Float16* Psw = Ps[wave];
  float m_r[8], l_r[8];
  v8f O[4];
#pragma unroll
  for (int j = 0; j < 8; ++j) { m_r[j] = -3.0e38f; l_r[j] = 0.f; }
#pragma unroll
  for (int t = 0; t < 4; ++t) O[t] = zero8();
  for (int ch = 0; ch < S_ / 32; ++ch) {
    for (int task = tid; task < 256; task += 128) {
      int key = task >> 3, seg = task & 7;
      int kp = ch * 32 + key;
      const _Float16* src = kk + ((size_t)b * S_ + kp) * D_ + h * DH_ + seg * 8;
#if ASYNC_LDS
      async_cp16(src, &Ks[key][seg * 8]);
#else
      *reinterpret_cast<uint4*>(&Ks[key][seg * 8]) = *reinterpret_cast<const uint4*>(src);
#endif
    }
    for (int task = tid; task < 256; task += 128) {
      int key = task >> 3, seg = task & 7;
      int kp = ch * 32 + key;
      uint4 val = *reinterpret_cast<const uint4*>(vv + ((size_t)b * S_ + kp) * D_ + h * DH_ + seg * 8);
      const _Float16* e = reinterpret_cast<const _Float16*>(&val);
#pragma unroll
      for (int j = 0; j < 8; ++j) Vt[seg * 8 + j][key] = e[j];
    }
#if ASYNC_LDS
    async_wait0();
#endif
    __syncthreads();
    v8f s0 = zero8(), s1 = zero8();
    {
      const _Float16* kr0 = &Ks[lm][half * 8];
      const _Float16* kr1 = &Ks[16 + lm][half * 8];
      v16h kf00 = ldfrag(kr0, kr0 + 16);
      v16h kf01 = ldfrag(kr0 + 32, kr0 + 48);
      v16h kf10 = ldfrag(kr1, kr1 + 16);
      v16h kf11 = ldfrag(kr1 + 32, kr1 + 48);
      s0 = wmma32(qf0, kf00, s0); s0 = wmma32(qf1, kf01, s0);
      s1 = wmma32(qf0, kf10, s1); s1 = wmma32(qf1, kf11, s1);
    }
    int kp0 = ch * 32 + lm, kp1 = kp0 + 16;
    bool ok0 = am[b * S_ + kp0] > 0;
    bool ok1 = am[b * S_ + kp1] > 0;
#pragma unroll
    for (int j = 0; j < 8; ++j) {
      int row = j + half * 8;
      float sv0 = ok0 ? s0[j] * scale : NEGV;
      float sv1 = ok1 ? s1[j] * scale : NEGV;
      float mx = fmaxf(sv0, sv1);
      mx = fmaxf(mx, shflxor(mx, 1));
      mx = fmaxf(mx, shflxor(mx, 2));
      mx = fmaxf(mx, shflxor(mx, 4));
      mx = fmaxf(mx, shflxor(mx, 8));
      float mnew = fmaxf(m_r[j], mx);
      float fct = __expf(m_r[j] - mnew);
      m_r[j] = mnew;
      float p0 = __expf(sv0 - mnew), p1 = __expf(sv1 - mnew);
      float rs = p0 + p1;
      rs += shflxor(rs, 1); rs += shflxor(rs, 2); rs += shflxor(rs, 4); rs += shflxor(rs, 8);
      l_r[j] = l_r[j] * fct + rs;
      O[0][j] *= fct; O[1][j] *= fct; O[2][j] *= fct; O[3][j] *= fct;
      Psw[row * 32 + lm]      = (_Float16)p0;
      Psw[row * 32 + 16 + lm] = (_Float16)p1;
    }
    asm volatile("s_wait_dscnt 0" ::: "memory");
    {
      const _Float16* pr = Psw + lm * 32 + half * 8;
      v16h pf = ldfrag(pr, pr + 16);
#pragma unroll
      for (int tn = 0; tn < 4; ++tn) {
        const _Float16* vr = &Vt[tn * 16 + lm][half * 8];
        v16h vf = ldfrag(vr, vr + 16);
        O[tn] = wmma32(pf, vf, O[tn]);
      }
    }
    __syncthreads();
  }
#pragma unroll
  for (int j = 0; j < 8; ++j) {
    int row = j + half * 8;
    int g = wave * 16 + row;
    float inv = 1.f / l_r[j];
#pragma unroll
    for (int tn = 0; tn < 4; ++tn)
      og[((size_t)b * G_ + g) * D_ + h * DH_ + tn * 16 + lm] = O[tn][j] * inv;
  }
}

// ---------------- gather rows of x at idx_g -> f16 ----------------
__global__ void k_gather_xg(const float* __restrict__ x, const int* __restrict__ idx_g,
                            _Float16* __restrict__ xg) {
  int g = blockIdx.x, b = blockIdx.y;
  int s = idx_g[b * G_ + g];
  for (int d = threadIdx.x; d < D_; d += blockDim.x)
    xg[((size_t)b * G_ + g) * D_ + d] = (_Float16)x[((size_t)b * S_ + s) * D_ + d];
}

// ---------------- scatter og rows into out at idx_g ----------------
__global__ void k_scatter(const float* __restrict__ og, const int* __restrict__ idx_g,
                          float* __restrict__ out) {
  int g = blockIdx.x, b = blockIdx.y;
  int s = idx_g[b * G_ + g];
  for (int d = threadIdx.x; d < D_; d += blockDim.x)
    out[((size_t)b * S_ + s) * D_ + d] = og[((size_t)b * G_ + g) * D_ + d];
}

// ---------------- classifier head ----------------
__global__ void k_classifier(const float* __restrict__ x, const float* __restrict__ Wc,
                             const float* __restrict__ bc, const int* __restrict__ lop,
                             const int* __restrict__ rop, float* __restrict__ out) {
  int lo = lop[0], ro = rop[0];
  int rows = NSEPS_ - lo - ro;
  int i = blockIdx.x * blockDim.x + threadIdx.x;
  if (i >= B_ * rows * C_) return;
  int c = i % C_;
  int r = i / C_;
  int b = r / rows;
  int t = lo + (r % rows);
  const float* xr = x + ((size_t)b * S_ + t) * D_;
  float s = bc[c];
  for (int d = 0; d < D_; ++d) s += xr[d] * Wc[d * C_ + c];
  out[i] = s;
}

// ---------------- host orchestration ----------------
extern "C" void kernel_launch(void* const* d_in, const int* in_sizes, int n_in,
                              void* d_out, int out_size, void* d_ws, size_t ws_size,
                              hipStream_t stream) {
  (void)in_sizes; (void)n_in; (void)out_size; (void)ws_size;
  const float* emb_tok = (const float*)d_in[0];
  const float* emb_pos = (const float*)d_in[1];
  const float* ln_e_s  = (const float*)d_in[2];
  const float* ln_e_b  = (const float*)d_in[3];
  const float* Wq  = (const float*)d_in[4];
  const float* bq  = (const float*)d_in[5];
  const float* Wk  = (const float*)d_in[6];
  const float* bk  = (const float*)d_in[7];
  const float* Wv  = (const float*)d_in[8];
  const float* bv  = (const float*)d_in[9];
  const float* Wqg = (const float*)d_in[10];
  const float* bqg = (const float*)d_in[11];
  const float* Wkg = (const float*)d_in[12];
  const float* bkg = (const float*)d_in[13];
  const float* Wvg = (const float*)d_in[14];
  const float* bvg = (const float*)d_in[15];
  const float* Wo  = (const float*)d_in[16];
  const float* bo  = (const float*)d_in[17];
  const float* ln1_s = (const float*)d_in[18];
  const float* ln1_b = (const float*)d_in[19];
  const float* Wf1 = (const float*)d_in[20];
  const float* bf1 = (const float*)d_in[21];
  const float* Wf2 = (const float*)d_in[22];
  const float* bf2 = (const float*)d_in[23];
  const float* ln2_s = (const float*)d_in[24];
  const float* ln2_b = (const float*)d_in[25];
  const float* Wc = (const float*)d_in[26];
  const float* bc = (const float*)d_in[27];
  const int* input_ids = (const int*)d_in[28];
  const int* attn_mask = (const int*)d_in[29];
  const int* lop = (const int*)d_in[30];
  const int* rop = (const int*)d_in[31];
  float* out = (float*)d_out;

  char* wsp = (char*)d_ws;
  size_t off = 0;
  auto alloc = [&](size_t bytes) -> char* {
    char* p = wsp + off;
    off += (bytes + 255) & ~(size_t)255;
    return p;
  };
  float*    x_f    = (float*)   alloc((size_t)MTOK * D_ * 4);
  _Float16* x_h    = (_Float16*)alloc((size_t)MTOK * D_ * 2);
  _Float16* q_h    = (_Float16*)alloc((size_t)MTOK * D_ * 2);
  _Float16* k_h    = (_Float16*)alloc((size_t)MTOK * D_ * 2);
  _Float16* v_h    = (_Float16*)alloc((size_t)MTOK * D_ * 2);
  _Float16* kga_h  = (_Float16*)alloc((size_t)MTOK * D_ * 2);
  _Float16* vga_h  = (_Float16*)alloc((size_t)MTOK * D_ * 2);
  _Float16* oat_h  = (_Float16*)alloc((size_t)MTOK * D_ * 2);
  _Float16* ffh_h  = (_Float16*)alloc((size_t)MTOK * FF_ * 2);
  float*    out_f  = (float*)   alloc((size_t)MTOK * D_ * 4);
  float*    proj_f = (float*)   alloc((size_t)MTOK * D_ * 4);
  _Float16* xg_h   = (_Float16*)alloc((size_t)B_ * G_ * D_ * 2);
  _Float16* qg_h   = (_Float16*)alloc((size_t)B_ * G_ * D_ * 2);
  float*    og_f   = (float*)   alloc((size_t)B_ * G_ * D_ * 4);
  int*      idxg   = (int*)     alloc((size_t)B_ * G_ * 4);
  int*      klo    = (int*)     alloc((size_t)B_ * S_ * 4);
  _Float16* Wq_h   = (_Float16*)alloc((size_t)L_ * D_ * D_ * 2);
  _Float16* Wk_h   = (_Float16*)alloc((size_t)L_ * D_ * D_ * 2);
  _Float16* Wv_h   = (_Float16*)alloc((size_t)L_ * D_ * D_ * 2);
  _Float16* Wqg_h  = (_Float16*)alloc((size_t)L_ * D_ * D_ * 2);
  _Float16* Wkg_h  = (_Float16*)alloc((size_t)L_ * D_ * D_ * 2);
  _Float16* Wvg_h  = (_Float16*)alloc((size_t)L_ * D_ * D_ * 2);
  _Float16* Wo_h   = (_Float16*)alloc((size_t)L_ * D_ * D_ * 2);
  _Float16* Wf1_h  = (_Float16*)alloc((size_t)L_ * D_ * FF_ * 2);
  _Float16* Wf2_h  = (_Float16*)alloc((size_t)L_ * FF_ * D_ * 2);

  auto conv = [&](const float* s, _Float16* dst, long long n) {
    unsigned blocks = (unsigned)((n + 255) / 256);
    k_f2h<<<blocks, 256, 0, stream>>>(s, dst, n);
  };
  conv(Wq,  Wq_h,  (long long)L_ * D_ * D_);
  conv(Wk,  Wk_h,  (long long)L_ * D_ * D_);
  conv(Wv,  Wv_h,  (long long)L_ * D_ * D_);
  conv(Wqg, Wqg_h, (long long)L_ * D_ * D_);
  conv(Wkg, Wkg_h, (long long)L_ * D_ * D_);
  conv(Wvg, Wvg_h, (long long)L_ * D_ * D_);
  conv(Wo,  Wo_h,  (long long)L_ * D_ * D_);
  conv(Wf1, Wf1_h, (long long)L_ * D_ * FF_);
  conv(Wf2, Wf2_h, (long long)L_ * FF_ * D_);

  k_globals<<<B_, 256, 0, stream>>>(input_ids, attn_mask, idxg, klo);
  k_embed_ln<<<MTOK, 256, 0, stream>>>(emb_tok, emb_pos, ln_e_s, ln_e_b, input_ids, x_f, x_h);

  dim3 gD(D_ / 64, MTOK / 128);        // 768-wide GEMMs over all tokens
  dim3 gQ(D_ / 64, (B_ * G_) / 128);   // global-query projection
  dim3 gF1(FF_ / 64, MTOK / 128);      // FFN up-projection

  for (int l = 0; l < L_; ++l) {
    size_t wo  = (size_t)l * D_ * D_;
    size_t bo_ = (size_t)l * D_;
    k_gemm<<<gD, 256, 0, stream>>>(x_h, Wq_h + wo,  bq + bo_,  nullptr, q_h,   MTOK, D_, D_, 0);
    k_gemm<<<gD, 256, 0, stream>>>(x_h, Wk_h + wo,  bk + bo_,  nullptr, k_h,   MTOK, D_, D_, 0);
    k_gemm<<<gD, 256, 0, stream>>>(x_h, Wv_h + wo,  bv + bo_,  nullptr, v_h,   MTOK, D_, D_, 0);
    k_gemm<<<gD, 256, 0, stream>>>(x_h, Wkg_h + wo, bkg + bo_, nullptr, kga_h, MTOK, D_, D_, 0);
    k_gemm<<<gD, 256, 0, stream>>>(x_h, Wvg_h + wo, bvg + bo_, nullptr, vga_h, MTOK, D_, D_, 0);
    k_gather_xg<<<dim3(G_, B_), 256, 0, stream>>>(x_f, idxg, xg_h);
    k_gemm<<<gQ, 256, 0, stream>>>(xg_h, Wqg_h + wo, bqg + bo_, nullptr, qg_h, B_ * G_, D_, D_, 0);
    k_band_attn<<<dim3(H_, NB_, B_), 512, 0, stream>>>(q_h, k_h, v_h, idxg, klo, out_f);
    k_glob_attn<<<dim3(H_, B_), 128, 0, stream>>>(qg_h, kga_h, vga_h, attn_mask, og_f);
    k_scatter<<<dim3(G_, B_), 256, 0, stream>>>(og_f, idxg, out_f);
    conv(out_f, oat_h, (long long)MTOK * D_);
    k_gemm<<<gD, 256, 0, stream>>>(oat_h, Wo_h + wo, bo + bo_, proj_f, nullptr, MTOK, D_, D_, 0);
    k_ln_resid<<<MTOK, 256, 0, stream>>>(x_f, proj_f, ln1_s + bo_, ln1_b + bo_, x_f, x_h);
    k_gemm<<<gF1, 256, 0, stream>>>(x_h, Wf1_h + (size_t)l * D_ * FF_, bf1 + (size_t)l * FF_,
                                    nullptr, ffh_h, MTOK, FF_, D_, 1);
    k_gemm<<<gD, 256, 0, stream>>>(ffh_h, Wf2_h + (size_t)l * FF_ * D_, bf2 + bo_,
                                   proj_f, nullptr, MTOK, D_, FF_, 0);
    k_ln_resid<<<MTOK, 256, 0, stream>>>(x_f, proj_f, ln2_s + bo_, ln2_b + bo_, x_f, x_h);
  }
  k_classifier<<<(B_ * NSEPS_ * C_ + 255) / 256, 256, 0, stream>>>(x_f, Wc, bc, lop, rop, out);
}
